// RNN_85521388798684
// MI455X (gfx1250) — matrix operand order repaired and checked by reading, not compiled, not run on hardware
//
#include <hip/hip_runtime.h>
#include <hip/hip_bf16.h>
#include <math.h>

// Problem sizes (fixed by reference)
#define D_ 512   // input size  (K of phase 1)
#define H_ 512   // hidden size (N of both GEMMs, K of recurrence)
#define N_ 64    // batch       (M of recurrence)
#define T_ 512   // timesteps
#define NWG 16   // workgroups in the scan kernel (co-resident, grid barrier)

typedef __attribute__((ext_vector_type(16))) __bf16 v16bf;
typedef __attribute__((ext_vector_type(2)))  __bf16 v2bf;
typedef __attribute__((ext_vector_type(8)))  float  v8f;
typedef __attribute__((ext_vector_type(4)))  unsigned int u32x4;
typedef __attribute__((ext_vector_type(8)))  int          i32x8;
typedef __attribute__((ext_vector_type(4)))  int          i32x4;

union FragAB { v16bf v; unsigned u[8]; };

// --- capability probes ---
#if defined(__has_builtin)
#  if __has_builtin(__builtin_amdgcn_tensor_load_to_lds)
#    define HAVE_TDM_BUILTIN 1
#  endif
#  if __has_builtin(__builtin_amdgcn_global_load_async_to_lds_b128)
#    define HAVE_ASYNC_B128 1
#  endif
#  if __has_builtin(__builtin_amdgcn_s_wait_asynccnt)
#    define HAVE_WAIT_ASYNC 1
#  endif
#  if __has_builtin(__builtin_amdgcn_cvt_pk_bf16_f32)
#    define HAVE_CVT_PK_BF16 1
#  endif
#endif
#if defined(__has_include)
#  if __has_include(<hip/amd_detail/amd_gfx1250_TDM.h>)
#    define TDM_SIX_ARG 1
#  endif
#endif

// Report which staging path the DEVICE pass selected (visible in stderr).
#if defined(__AMDGCN__)
#  if defined(HAVE_TDM_BUILTIN)
#    warning "gfx1250 device pass: scan uses TDM tensor_load_to_lds"
#  elif defined(HAVE_ASYNC_B128)
#    warning "gfx1250 device pass: scan uses global_load_async_to_lds_b128"
#  else
#    warning "gfx1250 device pass: scan uses plain cooperative copy"
#  endif
#  if defined(HAVE_CVT_PK_BF16)
#    warning "gfx1250 device pass: using v_cvt_pk_bf16_f32"
#  endif
#endif

// fp32 pair -> packed bf16 (RNE)
__device__ __forceinline__ unsigned short f2bf(float f) {
    unsigned u = __float_as_uint(f);
    u += 0x7FFFu + ((u >> 16) & 1u);
    return (unsigned short)(u >> 16);
}
__device__ __forceinline__ unsigned pack2(float lo, float hi) {
#if defined(HAVE_CVT_PK_BF16)
    v2bf r = __builtin_amdgcn_cvt_pk_bf16_f32(lo, hi);
    return __builtin_bit_cast(unsigned, r);
#else
    return (unsigned)f2bf(lo) | ((unsigned)f2bf(hi) << 16);
#endif
}

// ---------------------------------------------------------------------------
// Phase 0: convert h0 -> bf16 ping-pong buffer, zero grid-barrier state
// ---------------------------------------------------------------------------
__global__ void rnn_init_kernel(const float* __restrict__ h0,
                                unsigned short* __restrict__ hbuf,
                                unsigned* __restrict__ bar) {
    int i = blockIdx.x * blockDim.x + threadIdx.x;
    if (i < N_ * H_) hbuf[i] = f2bf(h0[i]);
    if (i == 0) { bar[0] = 0u; bar[1] = 0u; }
}

// ---------------------------------------------------------------------------
// Phase 1: xWx[(n*T+t), h] = sum_d x[(n*T+t), d] * Wx[d, h] + b[h]
// M = 32768, K = 512, N = 512. 128x128 block tile, BK=32, 8 waves,
// register-staged double-buffered LDS to overlap HBM loads with WMMA.
// ---------------------------------------------------------------------------
__global__ __launch_bounds__(256)
void gemm_xwx_kernel(const float* __restrict__ x,
                     const float* __restrict__ Wx,
                     const float* __restrict__ bias,
                     float* __restrict__ xwx) {
    __shared__ unsigned As[2][128][16];   // 2 x (128 rows x 32 K bf16)
    __shared__ unsigned Bs[2][128][16];   // 2 x (128 cols x 32 K bf16, transposed)

    const int m0   = blockIdx.x * 128;
    const int n0   = blockIdx.y * 128;
    const int tid  = threadIdx.x;
    const int lane = tid & 31;
    const int wv   = tid >> 5;      // 0..7
    const int waveM = wv & 3;       // 4 groups of 32 rows
    const int waveN = wv >> 2;      // 2 groups of 64 cols
    const int hi   = lane >> 4;
    const int l15  = lane & 15;

    v8f acc[2][4];
    #pragma unroll
    for (int mt = 0; mt < 2; ++mt)
        #pragma unroll
        for (int nt = 0; nt < 4; ++nt) {
            float bb = bias[n0 + waveN * 64 + nt * 16 + l15];
            #pragma unroll
            for (int e = 0; e < 8; ++e) acc[mt][nt][e] = bb;
        }

    const int ar = tid >> 1;        // A row 0..127
    const int ah = tid & 1;         // K half
    const int bc = tid & 127;       // B col 0..127
    const int bg = tid >> 7;        // K-pair group
    const float* xrow = x + (size_t)(m0 + ar) * D_ + ah * 16;
    const float* wcol = Wx + n0 + bc;

    float ax[16], bx[16];
    auto stage = [&](int k0) {      // global -> registers (loads in flight)
        const float4* xp = (const float4*)(xrow + k0);
        #pragma unroll
        for (int q = 0; q < 4; ++q) {
            float4 t4 = xp[q];
            ax[4*q+0] = t4.x; ax[4*q+1] = t4.y; ax[4*q+2] = t4.z; ax[4*q+3] = t4.w;
        }
        #pragma unroll
        for (int i = 0; i < 8; ++i) {
            int j = bg * 8 + i;
            bx[2*i]   = wcol[(size_t)(k0 + 2*j)     * H_];
            bx[2*i+1] = wcol[(size_t)(k0 + 2*j + 1) * H_];
        }
    };
    auto commit = [&](int buf) {    // registers -> bf16 LDS (v_cvt_pk_bf16_f32)
        #pragma unroll
        for (int i = 0; i < 8; ++i) As[buf][ar][ah*8 + i] = pack2(ax[2*i], ax[2*i+1]);
        #pragma unroll
        for (int i = 0; i < 8; ++i) Bs[buf][bc][bg*8 + i] = pack2(bx[2*i], bx[2*i+1]);
    };

    stage(0); commit(0);
    __syncthreads();

    for (int kt = 0; kt < 16; ++kt) {
        const int cur = kt & 1;
        if (kt < 15) stage((kt + 1) * 32);      // overlap next tile's HBM loads

        FragAB a[2], bfr[4];
        #pragma unroll
        for (int mt = 0; mt < 2; ++mt) {
            const unsigned* Au = &As[cur][waveM * 32 + mt * 16 + l15][0];
            #pragma unroll
            for (int j = 0; j < 4; ++j) {
                a[mt].u[j]     = Au[hi * 8 + j];       // K {0..7} | {16..23}
                a[mt].u[4 + j] = Au[hi * 8 + 4 + j];   // K {8..15} | {24..31}
            }
        }
        #pragma unroll
        for (int nt = 0; nt < 4; ++nt) {
            const unsigned* Bu = &Bs[cur][waveN * 64 + nt * 16 + l15][0];
            #pragma unroll
            for (int j = 0; j < 8; ++j) bfr[nt].u[j] = Bu[hi * 8 + j];
        }
        #pragma unroll
        for (int mt = 0; mt < 2; ++mt)
            #pragma unroll
            for (int nt = 0; nt < 4; ++nt)
                acc[mt][nt] = __builtin_amdgcn_wmma_f32_16x16x32_bf16(
                    false, a[mt].v, false, bfr[nt].v,
                    (short)0, acc[mt][nt], false, false);

        if (kt < 15) commit(cur ^ 1);
        __syncthreads();
    }

    #pragma unroll
    for (int mt = 0; mt < 2; ++mt)
        #pragma unroll
        for (int nt = 0; nt < 4; ++nt) {
            int row0 = m0 + waveM * 32 + mt * 16 + hi * 8;
            int col  = n0 + waveN * 64 + nt * 16 + l15;
            #pragma unroll
            for (int r = 0; r < 8; ++r)
                xwx[(size_t)(row0 + r) * H_ + col] = acc[mt][nt][r];
        }
}

// ---------------------------------------------------------------------------
// Phase 2: sequential scan  h_t = tanh(h_{t-1} @ Wh + xWx_t)
// 16 co-resident workgroups, Wh slice resident in LDS (32 KB).  Per step the
// 64 KB bf16 h-state tile is staged into LDS by the Tensor Data Mover (or
// async global->LDS copies), then consumed as ds_load_b128 WMMA fragments.
// One atomic grid barrier per timestep.
// ---------------------------------------------------------------------------
__global__ __launch_bounds__(256)
void rnn_scan_kernel(const float* __restrict__ Wh,
                     const float* __restrict__ xwx,
                     unsigned short* __restrict__ hbuf,   // 2 * (64*512) bf16
                     float* __restrict__ out,
                     unsigned* __restrict__ bar) {
    __shared__ unsigned Whs[32][256];   // 32 cols x 512 K (bf16 pairs), persistent
    __shared__ unsigned Hs[64 * 256];   // h_{t-1}: 64 rows x 512 K bf16 (64 KB)

    const int wg   = blockIdx.x;
    const int tid  = threadIdx.x;
    const int lane = tid & 31;
    const int wv   = tid >> 5;     // 0..7
    const int hi   = lane >> 4;
    const int l15  = lane & 15;
    const int mt   = wv & 3;       // M subtile
    const int nt   = wv >> 2;      // N subtile within 32-col slice

    // ---- preload Wh slice, transposed to B-fragment layout ----
    {
        int c   = tid & 31;
        int seg = tid >> 5;
        int col = wg * 32 + c;
        for (int i = 0; i < 32; ++i) {
            int j = seg * 32 + i;
            Whs[c][j] = pack2(Wh[(size_t)(2 * j)     * H_ + col],
                              Wh[(size_t)(2 * j + 1) * H_ + col]);
        }
    }

#if defined(HAVE_TDM_BUILTIN)
    // ---- TDM D# descriptors: 1-D tile, 32768 x 2-byte elements -> Hs ----
    const unsigned lds_off = (unsigned)(size_t)(void*)&Hs[0];
    u32x4 g0d[2];
    for (int bsel = 0; bsel < 2; ++bsel) {
        unsigned long long ga =
            (unsigned long long)(size_t)(hbuf + (size_t)bsel * (N_ * H_));
        u32x4 g0;
        g0.x = 1u;                                             // count=1, user D#
        g0.y = lds_off;                                        // LDS byte address
        g0.z = (unsigned)(ga & 0xffffffffu);                   // global_addr[31:0]
        g0.w = (unsigned)((ga >> 32) & 0x01ffffffu) | (2u << 30); // [56:32] | type=2
        g0d[bsel] = g0;
    }
    i32x8 g1;
    g1[0] = (int)(1u << 16);        // data_size = 2 bytes, no multicast/iterate/pad
    g1[1] = (int)(0x8000u << 16);   // tensor_dim0 = 32768 (low 16 bits << 16)
    g1[2] = 0;                      // tensor_dim0 hi / tensor_dim1 lo
    g1[3] = (int)(0x8000u << 16);   // tile_dim0 = 32768
    g1[4] = 0;                      // tile_dim1 / tile_dim2 unused
    g1[5] = 32768;                  // tensor_dim0_stride[31:0]
    g1[6] = 0;
    g1[7] = 0;
    i32x4 z4 = {0, 0, 0, 0};
#if defined(TDM_SIX_ARG)
    i32x8 z8 = {0, 0, 0, 0, 0, 0, 0, 0};
#endif
#endif

    __syncthreads();

    const int mrow = mt * 16 + l15;            // A row this lane reads
    const int ccol = nt * 16 + l15;            // B col (within slice)
    const int hcol = wg * 32 + nt * 16 + l15;  // global hidden column

    for (int t = 0; t < T_; ++t) {
        unsigned short* Hn = hbuf + (size_t)((t + 1) & 1) * (N_ * H_);

        // ---- stage h_{t-1} tile into LDS ----
#if defined(HAVE_TDM_BUILTIN)
        if (wv == 0) {
#if defined(TDM_SIX_ARG)
            __builtin_amdgcn_tensor_load_to_lds(g0d[t & 1], g1, z4, z4, z8, 0);
#else
            __builtin_amdgcn_tensor_load_to_lds(g0d[t & 1], g1, z4, z4, 0);
#endif
            asm volatile("s_wait_tensorcnt 0x0" ::: "memory");
        }
#elif defined(HAVE_ASYNC_B128)
        {
            const char* srcb =
                (const char*)(hbuf + (size_t)(t & 1) * (N_ * H_)) + tid * 16;
            char* dstb = (char*)&Hs[0] + tid * 16;
            #pragma unroll
            for (int i = 0; i < 16; ++i) {
                __builtin_amdgcn_global_load_async_to_lds_b128(
                    (const __attribute__((address_space(1))) void*)(srcb + i * 4096),
                    (__attribute__((address_space(3))) void*)(dstb + i * 4096),
                    0, 0);
            }
#if defined(HAVE_WAIT_ASYNC)
            __builtin_amdgcn_s_wait_asynccnt(0);
#else
            asm volatile("s_wait_asynccnt 0x0" ::: "memory");
#endif
        }
#else
        {
            const uint4* src = (const uint4*)(hbuf + (size_t)(t & 1) * (N_ * H_));
            uint4* dst = (uint4*)&Hs[0];
            #pragma unroll
            for (int i = 0; i < 16; ++i) dst[tid + 256 * i] = src[tid + 256 * i];
        }
#endif
        __syncthreads();

        v8f acc;
        #pragma unroll
        for (int e = 0; e < 8; ++e) acc[e] = 0.f;

        #pragma unroll 8
        for (int kk = 0; kk < 16; ++kk) {      // K = 512 in chunks of 32
            FragAB a, bf;
            const unsigned* Ar = &Hs[mrow * 256 + kk * 16];
            #pragma unroll
            for (int j = 0; j < 4; ++j) {
                a.u[j]     = Ar[hi * 8 + j];
                a.u[4 + j] = Ar[hi * 8 + 4 + j];
            }
            #pragma unroll
            for (int j = 0; j < 8; ++j) bf.u[j] = Whs[ccol][kk * 16 + hi * 8 + j];
            acc = __builtin_amdgcn_wmma_f32_16x16x32_bf16(
                false, a.v, false, bf.v, (short)0, acc, false, false);
        }

        // ---- epilogue: + xWx, tanh, write fp32 out and bf16 next state ----
        #pragma unroll
        for (int r = 0; r < 8; ++r) {
            int m = mt * 16 + hi * 8 + r;                       // batch row
            size_t idx = ((size_t)m * T_ + t) * H_ + hcol;      // (n, t, h)
            float v  = acc[r] + xwx[idx];
            float hv = tanhf(v);
            out[idx] = hv;
            Hn[(size_t)m * H_ + hcol] = f2bf(hv);
        }

        // ---- grid barrier (sense-reversing, atomic) ----
        __syncthreads();
        if (tid == 0) {
            __threadfence();
            unsigned gen  = atomicAdd(&bar[1], 0u);   // atomic load of generation
            unsigned prev = atomicAdd(&bar[0], 1u);
            if (prev == (unsigned)(NWG - 1)) {
                atomicExch(&bar[0], 0u);
                __threadfence();
                atomicAdd(&bar[1], 1u);               // release
            } else {
                while (atomicAdd(&bar[1], 0u) == gen) {
                    __builtin_amdgcn_s_sleep(1);
                }
            }
            __threadfence();
        }
        __syncthreads();
    }
}

// ---------------------------------------------------------------------------
extern "C" void kernel_launch(void* const* d_in, const int* in_sizes, int n_in,
                              void* d_out, int out_size, void* d_ws, size_t ws_size,
                              hipStream_t stream) {
    const float* x  = (const float*)d_in[0];   // (N, T, D)
    const float* h0 = (const float*)d_in[1];   // (N, H)
    const float* Wx = (const float*)d_in[2];   // (D, H)
    const float* Wh = (const float*)d_in[3];   // (H, H)
    const float* b  = (const float*)d_in[4];   // (H,)
    float* out = (float*)d_out;                // (N, T, H)

    // workspace layout: xWx fp32 (64 MB) | h bf16 ping-pong (2*64KB) | barrier
    float*          xwx  = (float*)d_ws;
    unsigned short* hbuf = (unsigned short*)(xwx + (size_t)N_ * T_ * H_);
    unsigned*       bar  = (unsigned*)(hbuf + 2 * (size_t)N_ * H_);

    rnn_init_kernel<<<dim3((N_ * H_ + 255) / 256), dim3(256), 0, stream>>>(h0, hbuf, bar);
    gemm_xwx_kernel<<<dim3((N_ * T_) / 128, H_ / 128), dim3(256), 0, stream>>>(x, Wx, b, xwx);
    rnn_scan_kernel<<<dim3(NWG), dim3(256), 0, stream>>>(Wh, xwx, hbuf, out, bar);
}
